// AugmentedLSTMTagger_69114613730083
// MI455X (gfx1250) — matrix operand order repaired and checked by reading, not compiled, not run on hardware
//
#include <hip/hip_runtime.h>

// Problem dims
#define B_  32
#define S_  1024
#define C_  12
#define E_  256
#define H_  512
#define CS_ 128
#define T_  50

typedef __attribute__((ext_vector_type(16))) __bf16 v16bf;
typedef __attribute__((ext_vector_type(8)))  float  v8f;

__device__ __forceinline__ float sigf(float x) { return 1.f / (1.f + __expf(-x)); }

__device__ __forceinline__ v8f v8f_zero() {
  v8f z;
#pragma unroll
  for (int e = 0; e < 8; ++e) z[e] = 0.f;
  return z;
}

// CDNA5 async copy: GLOBAL_LOAD_ASYNC_TO_LDS_B128 (GV mode).
// lds_off = LDS byte address (low 32 bits of generic LDS pointer), gaddr = 64-bit global.
__device__ __forceinline__ void async_copy_b128(unsigned int lds_off, const void* gaddr) {
  asm volatile("global_load_async_to_lds_b128 %0, %1, off"
               :: "v"(lds_off), "v"(gaddr)
               : "memory");
}
__device__ __forceinline__ void wait_async0() {
  asm volatile("s_wait_asynccnt 0x0" ::: "memory");
}

// A-matrix 16x32 bf16 fragment (ISA 7.12.2): lanes 0-15 row M=lane hold K {0..7,16..23},
// lanes 16-31 row M=lane-16 hold K {8..15,24..31}.
__device__ __forceinline__ v16bf load_a_frag(const __bf16* tile, int ldm, int lane) {
  union { v16bf v; uint4 q[2]; } u;
  const int m  = lane & 15;
  const int lk = (lane < 16) ? 0 : 8;
  const __bf16* p = tile + m * ldm + lk;
  u.q[0] = *(const uint4*)(p);        // K = lk .. lk+7
  u.q[1] = *(const uint4*)(p + 16);   // K = lk+16 .. lk+23
  return u.v;
}

// B-matrix 32x16 bf16 fragment: lane n (0-15) holds col n K {0..7,16..23}; lane n+16 holds K {8..15,24..31}.
// For out = X * W^T with W row-major [N][K]: lane reads K kb..kb+7 and kb+16..kb+23 from W row n.
__device__ __forceinline__ v16bf load_b_frag(const __bf16* wrow0, int ldk, int lane) {
  union { v16bf v; uint4 q[2]; } u;
  const int n  = lane & 15;
  const int kb = (lane < 16) ? 0 : 8;
  const __bf16* p = wrow0 + (size_t)n * ldk + kb;
  u.q[0] = *(const uint4*)(p);
  u.q[1] = *(const uint4*)(p + 16);
  return u.v;
}

__device__ __forceinline__ v8f wmma_bf16(v16bf a, v16bf b, v8f c) {
  return __builtin_amdgcn_wmma_f32_16x16x32_bf16(false, a, false, b, (short)0, c, false, false);
}

// ---------------- prep kernels ----------------

// dst[rows][kin+khid] bf16 = [Wih | Whh] so the LSTM step is one GEMM over [x|h]
__global__ void pack_lstm_weights(const float* __restrict__ Wih,
                                  const float* __restrict__ Whh,
                                  __bf16* __restrict__ dst,
                                  int rows, int kin, int khid) {
  const int K = kin + khid;
  const long long total = (long long)rows * K;
  long long i = (long long)blockIdx.x * blockDim.x + threadIdx.x;
  if (i >= total) return;
  int r = (int)(i / K), k = (int)(i % K);
  float v = (k < kin) ? Wih[(long long)r * kin + k]
                      : Whh[(long long)r * khid + (k - kin)];
  dst[i] = (__bf16)v;
}

__global__ void convert_bf16(const float* __restrict__ src, __bf16* __restrict__ dst, int n) {
  int i = blockIdx.x * blockDim.x + threadIdx.x;
  if (i < n) dst[i] = (__bf16)src[i];
}

// Wt [50][512] -> WtP [64][512] bf16, zero padded rows
__global__ void pack_wt(const float* __restrict__ Wt, __bf16* __restrict__ dst) {
  int i = blockIdx.x * blockDim.x + threadIdx.x;
  if (i >= 64 * 512) return;
  int r = i >> 9, k = i & 511;
  dst[i] = (__bf16)((r < T_) ? Wt[r * 512 + k] : 0.f);
}

// word_emb gather -> x_word[s][b][0:256] bf16  (x_word layout [S][B][2E])
__global__ void gather_word(const int* __restrict__ word_idx,
                            const float* __restrict__ wemb,
                            __bf16* __restrict__ x_word) {
  int i = blockIdx.x * blockDim.x + threadIdx.x;    // over B*S*E
  if (i >= B_ * S_ * E_) return;
  int w = i >> 8;              // word = b*S + s
  int c = i & 255;
  int b = w >> 10, s = w & 1023;
  float v = wemb[(size_t)word_idx[w] * E_ + c];
  x_word[((size_t)s * B_ + b) * (2 * E_) + c] = (__bf16)v;
}

// ---------------- char-level LSTM ----------------
// One WG per 16 words; 8 waves; 12 recurrent steps; gates[16][1024] via bf16 WMMA.
__global__ void __launch_bounds__(256)
char_lstm_kernel(const int* __restrict__ char_idx, const __bf16* __restrict__ cemb,
                 const __bf16* __restrict__ WcP, const float* __restrict__ bc,
                 __bf16* __restrict__ x_word) {
  extern __shared__ char smem[];
  __bf16* xh  = (__bf16*)smem;                        // [16][512]  x | h
  float*  gts = (float*)(smem + 16 * 512 * 2);        // [16][1024]
  float*  cst = (float*)((char*)gts + 16 * 1024 * 4); // [16][256]

  const int tid  = threadIdx.x;
  const int lane = tid & 31;
  const int wave = tid >> 5;
  const int row0 = blockIdx.x * 16;

  for (int i = tid; i < 16 * 256; i += 256) {         // h = 0, c = 0
    xh[(i >> 8) * 512 + 256 + (i & 255)] = (__bf16)0.f;
    cst[i] = 0.f;
  }
  __syncthreads();

  for (int t = 0; t < C_; ++t) {
    { // async-gather x_t: per-word char embedding row (256 bf16) -> LDS
      int r  = tid >> 4;
      int c0 = (tid & 15) * 16;
      int ci = char_idx[(row0 + r) * C_ + t];
      const __bf16* src = cemb + (size_t)ci * E_ + c0;
      unsigned int dst = (unsigned int)(uintptr_t)&xh[r * 512 + c0];
      async_copy_b128(dst,      src);
      async_copy_b128(dst + 16, src + 8);
      wait_async0();
    }
    __syncthreads();

    { // GEMM: wave -> gate cols [wave*128, wave*128+128), K = 512
      v8f acc[8];
#pragma unroll
      for (int i = 0; i < 8; ++i) acc[i] = v8f_zero();
      for (int kc = 0; kc < 16; ++kc) {
        v16bf a = load_a_frag(xh + kc * 32, 512, lane);
        const __bf16* wbase = WcP + (size_t)(wave * 128) * 512 + kc * 32;
        __builtin_prefetch((const void*)(wbase + 32), 0, 1);
#pragma unroll
        for (int nb = 0; nb < 8; ++nb) {
          v16bf b = load_b_frag(wbase + (size_t)nb * 16 * 512, 512, lane);
          acc[nb] = wmma_bf16(a, b, acc[nb]);
        }
      }
      const int rbase = (lane < 16) ? 0 : 8;
      const int nl = lane & 15;
#pragma unroll
      for (int nb = 0; nb < 8; ++nb) {
        int col = wave * 128 + nb * 16 + nl;
#pragma unroll
        for (int r = 0; r < 8; ++r) gts[(rbase + r) * 1024 + col] = acc[nb][r];
      }
    }
    __syncthreads();

    { // elementwise cell: row = tid>>4, 16 units per thread (packed b128 stores)
      int r  = tid >> 4;
      int u0 = (tid & 15) * 16;
      const float* g = gts + r * 1024;
      union { __bf16 h[16]; uint4 q[2]; } hb;
      for (int k = 0; k < 16; ++k) {
        int u = u0 + k;
        float gi = g[u]           + bc[u];
        float gf = g[E_ + u]      + bc[E_ + u];
        float gg = g[2 * E_ + u]  + bc[2 * E_ + u];
        float go = g[3 * E_ + u]  + bc[3 * E_ + u];
        float c = sigf(gf) * cst[r * E_ + u] + sigf(gi) * tanhf(gg);
        cst[r * E_ + u] = c;
        hb.h[k] = (__bf16)(sigf(go) * tanhf(c));
      }
      *(uint4*)&xh[r * 512 + 256 + u0]     = hb.q[0];
      *(uint4*)&xh[r * 512 + 256 + u0 + 8] = hb.q[1];
      if (t == C_ - 1) { // char_feat -> x_word[s][b][256:512]
        int w = row0 + r;
        int b = w >> 10, s = w & 1023;
        __bf16* dst = &x_word[((size_t)s * B_ + b) * (2 * E_) + 256 + u0];
        *(uint4*)(dst)     = hb.q[0];
        *(uint4*)(dst + 8) = hb.q[1];
      }
    }
    __syncthreads();
  }
}

// ---------------- word-level LSTM ----------------
// Single 1024-thread WG (32 waves) runs the 1024-step serial chain.
// gates[32][2048] f32 (256KB) + xh[32][1024] bf16 (64KB) = 320KB LDS.
__global__ void __launch_bounds__(1024)
word_lstm_kernel(const __bf16* __restrict__ x_word, const __bf16* __restrict__ WwP,
                 const float* __restrict__ bw, __bf16* __restrict__ h_out) {
  extern __shared__ char smem[];
  __bf16* xh  = (__bf16*)smem;                     // [32][1024]  x | h
  float*  gts = (float*)(smem + 32 * 1024 * 2);    // [32][2048]

  const int tid  = threadIdx.x;
  const int lane = tid & 31;
  const int wave = tid >> 5;

  float creg[16];                                  // cell state: row tid>>5, units lane*16..+15
#pragma unroll
  for (int i = 0; i < 16; ++i) creg[i] = 0.f;
  for (int i = tid; i < 32 * 512; i += 1024)
    xh[(i >> 9) * 1024 + 512 + (i & 511)] = (__bf16)0.f;
  __syncthreads();

  for (int s = 0; s < S_; ++s) {
    { // async-load x_s: 32 rows x 512 bf16 -> LDS
      int r  = tid >> 5;
      int c0 = (tid & 31) * 16;
      const __bf16* src = x_word + ((size_t)s * B_ + r) * 512 + c0;
      unsigned int dst = (unsigned int)(uintptr_t)&xh[r * 1024 + c0];
      async_copy_b128(dst,      src);
      async_copy_b128(dst + 16, src + 8);
      wait_async0();
    }
    __syncthreads();

    { // GEMM: wave -> gate cols [wave*64, +64), 2 m-tiles, K = 1024
      v8f acc[2][4];
#pragma unroll
      for (int m = 0; m < 2; ++m)
#pragma unroll
        for (int n = 0; n < 4; ++n) acc[m][n] = v8f_zero();
      for (int kc = 0; kc < 32; ++kc) {
        v16bf a0 = load_a_frag(xh + kc * 32, 1024, lane);
        v16bf a1 = load_a_frag(xh + 16 * 1024 + kc * 32, 1024, lane);
        const __bf16* wbase = WwP + (size_t)(wave * 64) * 1024 + kc * 32;
        __builtin_prefetch((const void*)(wbase + 32), 0, 1);
#pragma unroll
        for (int nb = 0; nb < 4; ++nb) {
          v16bf b = load_b_frag(wbase + (size_t)nb * 16 * 1024, 1024, lane);
          acc[0][nb] = wmma_bf16(a0, b, acc[0][nb]);
          acc[1][nb] = wmma_bf16(a1, b, acc[1][nb]);
        }
      }
      const int rbase = (lane < 16) ? 0 : 8;
      const int nl = lane & 15;
#pragma unroll
      for (int mt = 0; mt < 2; ++mt)
#pragma unroll
        for (int nb = 0; nb < 4; ++nb) {
          int col = wave * 64 + nb * 16 + nl;
#pragma unroll
          for (int r = 0; r < 8; ++r)
            gts[(mt * 16 + rbase + r) * 2048 + col] = acc[mt][nb][r];
        }
    }
    __syncthreads();

    { // elementwise cell: row = tid>>5, units lane*16..+15 (packed b128 stores)
      int r  = tid >> 5;
      int u0 = lane * 16;
      const float* g = gts + r * 2048;
      union { __bf16 h[16]; uint4 q[2]; } hb;
      for (int k = 0; k < 16; ++k) {
        int u = u0 + k;
        float gi = g[u]           + bw[u];
        float gf = g[H_ + u]      + bw[H_ + u];
        float gg = g[2 * H_ + u]  + bw[2 * H_ + u];
        float go = g[3 * H_ + u]  + bw[3 * H_ + u];
        float c = sigf(gf) * creg[k] + sigf(gi) * tanhf(gg);
        creg[k] = c;
        hb.h[k] = (__bf16)(sigf(go) * tanhf(c));
      }
      *(uint4*)&xh[r * 1024 + 512 + u0]     = hb.q[0];
      *(uint4*)&xh[r * 1024 + 512 + u0 + 8] = hb.q[1];
      __bf16* dst = &h_out[((size_t)s * B_ + r) * H_ + u0];   // [s][b][H]
      *(uint4*)(dst)     = hb.q[0];
      *(uint4*)(dst + 8) = hb.q[1];
    }
    __syncthreads();
  }
}

// ---------------- tag projection + log_softmax ----------------
// WG handles 128 tokens; logits [128][64] via WMMA (Wt padded to 64 rows), fused log-softmax.
__global__ void __launch_bounds__(256)
tag_kernel(const __bf16* __restrict__ h_all, const __bf16* __restrict__ WtP,
           const float* __restrict__ bt, float* __restrict__ out) {
  __shared__ float lg[128 * 64];
  const int tid  = threadIdx.x;
  const int lane = tid & 31;
  const int wave = tid >> 5;
  const int row0 = blockIdx.x * 128;

  { // wave -> m-tile `wave` (16 rows) x all 4 n-tiles, K = 512
    v8f acc[4];
#pragma unroll
    for (int n = 0; n < 4; ++n) acc[n] = v8f_zero();
    const __bf16* abase = h_all + (size_t)(row0 + wave * 16) * H_;
    for (int kc = 0; kc < 16; ++kc) {
      v16bf a = load_a_frag(abase + kc * 32, H_, lane);
#pragma unroll
      for (int nb = 0; nb < 4; ++nb) {
        v16bf b = load_b_frag(WtP + (size_t)nb * 16 * H_ + kc * 32, H_, lane);
        acc[nb] = wmma_bf16(a, b, acc[nb]);
      }
    }
    const int rbase = (lane < 16) ? 0 : 8;
    const int nl = lane & 15;
#pragma unroll
    for (int nb = 0; nb < 4; ++nb)
#pragma unroll
      for (int r = 0; r < 8; ++r)
        lg[(wave * 16 + rbase + r) * 64 + nb * 16 + nl] = acc[nb][r];
  }
  __syncthreads();

  if (tid < 128) {
    int token = row0 + tid;        // token = s*B + b
    int b = token & 31, s = token >> 5;
    float* row = lg + tid * 64;
    float mx = -1e30f;
    for (int t = 0; t < T_; ++t) {
      float v = row[t] + bt[t];
      row[t] = v;
      if (v > mx) mx = v;
    }
    float sum = 0.f;
    for (int t = 0; t < T_; ++t) sum += __expf(row[t] - mx);
    float lse = mx + __logf(sum);
    float* o = out + ((size_t)b * S_ + s) * T_;
    for (int t = 0; t < T_; ++t) o[t] = row[t] - lse;
  }
}

// ---------------- launch ----------------
extern "C" void kernel_launch(void* const* d_in, const int* in_sizes, int n_in,
                              void* d_out, int out_size, void* d_ws, size_t ws_size,
                              hipStream_t stream) {
  (void)in_sizes; (void)n_in; (void)out_size; (void)ws_size;
  const int*   char_idx = (const int*)d_in[0];
  const int*   word_idx = (const int*)d_in[1];
  const float* char_emb = (const float*)d_in[2];
  const float* word_emb = (const float*)d_in[3];
  const float* Wc_ih    = (const float*)d_in[4];
  const float* Wc_hh    = (const float*)d_in[5];
  const float* bc       = (const float*)d_in[6];
  const float* Ww_ih    = (const float*)d_in[7];
  const float* Ww_hh    = (const float*)d_in[8];
  const float* bw       = (const float*)d_in[9];
  const float* Wt       = (const float*)d_in[10];
  const float* bt       = (const float*)d_in[11];
  float* out = (float*)d_out;

  char* ws = (char*)d_ws;
  size_t off = 0;
  __bf16* WcP   = (__bf16*)(ws + off); off += (size_t)1024 * 512 * 2;       // 1 MB
  __bf16* WwP   = (__bf16*)(ws + off); off += (size_t)2048 * 1024 * 2;      // 4 MB
  __bf16* WtP   = (__bf16*)(ws + off); off += (size_t)64 * 512 * 2;
  __bf16* cembB = (__bf16*)(ws + off); off += (size_t)CS_ * E_ * 2;
  __bf16* xword = (__bf16*)(ws + off); off += (size_t)B_ * S_ * 2 * E_ * 2; // 32 MB
  __bf16* hall  = (__bf16*)(ws + off); off += (size_t)B_ * S_ * H_ * 2;     // 32 MB

  pack_lstm_weights<<<(1024 * 512 + 255) / 256, 256, 0, stream>>>(Wc_ih, Wc_hh, WcP, 1024, 256, 256);
  pack_lstm_weights<<<(2048 * 1024 + 255) / 256, 256, 0, stream>>>(Ww_ih, Ww_hh, WwP, 2048, 512, 512);
  convert_bf16<<<(CS_ * E_ + 255) / 256, 256, 0, stream>>>(char_emb, cembB, CS_ * E_);
  pack_wt<<<(64 * 512 + 255) / 256, 256, 0, stream>>>(Wt, WtP);
  gather_word<<<(B_ * S_ * E_ + 255) / 256, 256, 0, stream>>>(word_idx, word_emb, xword);

  // char LSTM: 16 KB xh + 64 KB gates + 16 KB cell = 96 KB dynamic LDS
  char_lstm_kernel<<<(B_ * S_) / 16, 256, 16 * 512 * 2 + 16 * 1024 * 4 + 16 * 256 * 4, stream>>>(
      char_idx, cembB, WcP, bc, xword);

  // word LSTM: 64 KB xh + 256 KB gates = 320 KB dynamic LDS (full CDNA5 WGP pool)
  word_lstm_kernel<<<1, 1024, 32 * 1024 * 2 + 32 * 2048 * 4, stream>>>(xword, WwP, bw, hall);

  tag_kernel<<<(B_ * S_) / 128, 256, 0, stream>>>(hall, WtP, bt, out);
}
